// SwarmTransformer_8005819040260
// MI455X (gfx1250) — matrix operand order repaired and checked by reading, not compiled, not run on hardware
//
#include <hip/hip_runtime.h>
#include <hip/hip_bf16.h>
#include <math.h>

// ---------------------------------------------------------------------------
// GPT-2 small forward on gfx1250 (MI455X), bf16 WMMA (f32 accumulate).
// Residual stream fp32; GEMM/attention operands bf16.
// GEMM: 64x128 tile, double-buffered LDS, async global->LDS A-tile copies.
// ---------------------------------------------------------------------------

typedef __attribute__((ext_vector_type(16))) __bf16 v16bf;
typedef __attribute__((ext_vector_type(8)))  float  v8f;
typedef __attribute__((ext_vector_type(4)))  int    v4i;

#define WMMA_BF16(A, B, C) \
  __builtin_amdgcn_wmma_f32_16x16x32_bf16(false, (A), false, (B), (short)0, (C), false, false)

// Async global->LDS (ASYNCcnt-tracked), device pass only; fallback otherwise.
#if defined(__gfx1250__) && __has_builtin(__builtin_amdgcn_global_load_async_to_lds_b128)
#define HAVE_ASYNC_LDS 1
#else
#define HAVE_ASYNC_LDS 0
#endif

#if HAVE_ASYNC_LDS
typedef __attribute__((address_space(1))) v4i as1_v4i;
typedef __attribute__((address_space(3))) v4i as3_v4i;
__device__ __forceinline__ void async_copy_b128(const void* g, void* l) {
  // Builtin prototype (probe-confirmed): (v4i addrspace(1)*, v4i addrspace(3)*, imm, imm)
  __builtin_amdgcn_global_load_async_to_lds_b128((as1_v4i*)g, (as3_v4i*)l, 0, 0);
}
__device__ __forceinline__ void wait_async0() {
#if __has_builtin(__builtin_amdgcn_s_wait_asynccnt)
  __builtin_amdgcn_s_wait_asynccnt(0);
#else
  asm volatile("s_wait_asynccnt 0x0" ::: "memory");
#endif
}
#endif

// A-frag 16x32 k order used here (not AMD's table order): lane l holds row m=l&15;
// lanes 0-15 K=0..15 (element e -> K=e), lanes 16-31 K=16..31 (matches B).
__device__ __forceinline__ int a_k_of(int lane, int e) {
  return ((lane & 16) ? 16 : 0) + e;
}
// B-frag 32x16 k order used here (not AMD's table order): lane l holds col n=l&15; lanes 0-15
// K=0..15 (element e -> K=e), lanes 16-31 K=16..31.
__device__ __forceinline__ int b_k_of(int lane, int e) {
  return ((lane & 16) ? 16 : 0) + e;
}

enum { GF_ADDRES = 1, GF_WF32 = 2, GF_GELU = 4, GF_WBF16 = 8 };

#define HDIM 768
#define NHEADS 12
#define HEADD 64

// ---------------------------------------------------------------------------
// Embedding
// ---------------------------------------------------------------------------
__global__ __launch_bounds__(256) void swarm_embed_kernel(
    const int* __restrict__ ids, const float* __restrict__ tok,
    const float* __restrict__ pos, float* __restrict__ x, int T) {
  long row = blockIdx.x;
  int t = (int)(row % T);
  long id = ids[row];
  const float* tr = tok + id * HDIM;
  const float* pr = pos + (long)t * HDIM;
  float* xr = x + row * HDIM;
  for (int h = threadIdx.x; h < HDIM; h += blockDim.x) xr[h] = tr[h] + pr[h];
}

// ---------------------------------------------------------------------------
// LayerNorm: one wave per token (wave32, 24 elems/lane), writes bf16.
// ---------------------------------------------------------------------------
__global__ __launch_bounds__(256) void swarm_ln_kernel(
    const float* __restrict__ x, const float* __restrict__ w,
    const float* __restrict__ b, __bf16* __restrict__ out, int M) {
  int lane = threadIdx.x & 31;
  int wv = threadIdx.x >> 5;
  long row = (long)blockIdx.x * 8 + wv;
  if (row >= M) return;
  const float* xr = x + row * HDIM;
  float vals[24];
  float s = 0.f;
#pragma unroll
  for (int i = 0; i < 24; ++i) { vals[i] = xr[lane + i * 32]; s += vals[i]; }
#pragma unroll
  for (int m = 1; m < 32; m <<= 1) s += __shfl_xor(s, m, 32);
  float mean = s * (1.f / (float)HDIM);
  float v = 0.f;
#pragma unroll
  for (int i = 0; i < 24; ++i) { float d = vals[i] - mean; v += d * d; }
#pragma unroll
  for (int m = 1; m < 32; m <<= 1) v += __shfl_xor(v, m, 32);
  float rstd = rsqrtf(v * (1.f / (float)HDIM) + 1e-5f);
  __bf16* orow = out + row * HDIM;
#pragma unroll
  for (int i = 0; i < 24; ++i) {
    int h = lane + i * 32;
    orow[h] = (__bf16)((vals[i] - mean) * rstd * w[h] + b[h]);
  }
}

// ---------------------------------------------------------------------------
// GEMM: C[M,N] = A[M,K](bf16) @ W[N,K](fp32 -> bf16 in LDS)^T + bias
// Block tile 64(M) x 128(N), K-step 32, 8 waves (4 row x 2 col), each wave
// computes 16x64 -> one A-frag reused over 4 B-frags -> 4 WMMAs per K-step.
// Double-buffered LDS; A-tile staged via GLOBAL_LOAD_ASYNC_TO_LDS_B128
// (one b128 per thread per K-step), W-tile software-pipelined via VGPRs.
// ---------------------------------------------------------------------------
__global__ __launch_bounds__(256) void swarm_gemm_kernel(
    const __bf16* __restrict__ A, const float* __restrict__ W,
    const float* __restrict__ bias, float* __restrict__ Cf,
    __bf16* __restrict__ Cb, int M, int N, int K, int flags) {
  __shared__ __bf16 sA[2][64][40];   // 80B row stride: 16B-aligned b128 dests
  __shared__ __bf16 sB[2][128][40];
  int tid = threadIdx.x;
  int lane = tid & 31;
  int wave = tid >> 5;
  int wm = (wave & 3) * 16;    // tile-local row base of this wave
  int wn = (wave >> 2) * 64;   // tile-local col base of this wave
  long rowBase = (long)blockIdx.y * 64;
  long colBase = (long)blockIdx.x * 128;

  // A cooperative load: 64x32 bf16 tile, one 16B chunk per thread.
  int la_r = (tid * 8) >> 5;         // 0..63
  int la_c = (tid * 8) & 31;         // 0,8,16,24
  const __bf16* Arow = A + (rowBase + la_r) * (long)K + la_c;
  // W cooperative load: 128x32 fp32 tile, 16 floats per thread.
  int lw_r = tid >> 1;               // 0..127
  int lw_c = (tid & 1) * 16;         // 0 or 16
  bool wok = (colBase + lw_r) < N;
  const float* Wrow = W + (wok ? (colBase + lw_r) : 0) * (long)K + lw_c;

  v8f acc[4] = {};

  auto issueA = [&](int k0, int buf) {
#if HAVE_ASYNC_LDS
    async_copy_b128(Arow + k0, &sA[buf][la_r][la_c]);
#else
    const __bf16* src = Arow + k0;
#pragma unroll
    for (int i = 0; i < 8; ++i) sA[buf][la_r][la_c + i] = src[i];
#endif
  };

  // Prologue: stage K-tile 0 into buffer 0.
  issueA(0, 0);
  {
    float wr[16];
#pragma unroll
    for (int i = 0; i < 16; ++i) wr[i] = wok ? Wrow[i] : 0.f;
#pragma unroll
    for (int i = 0; i < 16; ++i) sB[0][lw_r][lw_c + i] = (__bf16)wr[i];
  }

  int p = 0;
  for (int k0 = 0; k0 < K; k0 += 32) {
#if HAVE_ASYNC_LDS
    wait_async0();                  // own async A-tile landed in LDS
#endif
    __syncthreads();                // all waves' tiles visible

    bool hasNext = (k0 + 32) < K;
    float wr[16];
    if (hasNext) {
      issueA(k0 + 32, p ^ 1);       // async: overlaps with WMMAs below
#pragma unroll
      for (int i = 0; i < 16; ++i) wr[i] = wok ? Wrow[k0 + 32 + i] : 0.f;
    }

    // Fragments + 4 WMMAs from buffer p.
    v16bf af;
    int am = wm + (lane & 15);
#pragma unroll
    for (int e = 0; e < 16; ++e) af[e] = sA[p][am][a_k_of(lane, e)];
#pragma unroll
    for (int f = 0; f < 4; ++f) {
      v16bf bf;
      int bn = wn + f * 16 + (lane & 15);
#pragma unroll
      for (int e = 0; e < 16; ++e) bf[e] = sB[p][bn][b_k_of(lane, e)];
      acc[f] = WMMA_BF16(af, bf, acc[f]);
    }

    if (hasNext) {
#pragma unroll
      for (int i = 0; i < 16; ++i) sB[p ^ 1][lw_r][lw_c + i] = (__bf16)wr[i];
    }
    p ^= 1;
  }

  // Epilogue: C 16x16 f32 layout: lanes 0-15 rows r, lanes 16-31 rows r+8.
  int colLocal = lane & 15;
  int rowOff = (lane & 16) ? 8 : 0;
#pragma unroll
  for (int f = 0; f < 4; ++f) {
    long n = colBase + wn + f * 16 + colLocal;
    if (n >= N) continue;
    float bval = bias ? bias[n] : 0.f;
#pragma unroll
    for (int r = 0; r < 8; ++r) {
      long m = rowBase + wm + rowOff + r;
      if (m >= M) continue;
      float val = acc[f][r] + bval;
      if (flags & GF_GELU) val = 0.5f * val * (1.f + erff(val * 0.70710678118f));
      long idx = m * (long)N + n;
      if (flags & GF_ADDRES) {
        val += Cf[idx];
        Cf[idx] = val;
      } else if (flags & GF_WF32) {
        Cf[idx] = val;
      }
      if (flags & GF_WBF16) Cb[idx] = (__bf16)val;
    }
  }
}

// ---------------------------------------------------------------------------
// Flash attention: one wave per (b, head, 16-row q tile). 32-key chunks:
// S(16x32) = Q(16x64)K^T via 4 WMMAs, online softmax (causal), P staged via
// LDS (single-wave workgroup: barrier -> S_NOP) to A-frag layout, then
// O = O*corr + P(16x32) V(32x64) via 4 WMMAs.
// ---------------------------------------------------------------------------
__global__ __launch_bounds__(32) void swarm_attn_kernel(
    const __bf16* __restrict__ q, const __bf16* __restrict__ k,
    const __bf16* __restrict__ v, __bf16* __restrict__ o, int T) {
  __shared__ __bf16 sP[16][32];
  int lane = threadIdx.x;
  int q0 = blockIdx.x * 16;
  int head = blockIdx.y;
  long baseRow = (long)blockIdx.z * T;
  int col = head * HEADD;

  int m = lane & 15;
  int nloc = lane & 15;
  int rowOff = (lane & 16) ? 8 : 0;
  int kb = (lane & 16) ? 16 : 0;

  v16bf aq0, aq1;
  {
    const __bf16* qrow = q + (baseRow + q0 + m) * HDIM + col;
#pragma unroll
    for (int e = 0; e < 16; ++e) {
      int d = a_k_of(lane, e);
      aq0[e] = qrow[d];
      aq1[e] = qrow[32 + d];
    }
  }

  v8f oacc[4] = {};
  float run_m[8], run_l[8];
#pragma unroll
  for (int r = 0; r < 8; ++r) { run_m[r] = -3.0e38f; run_l[r] = 0.f; }

  int jmax = (q0 + 15) >> 5;
  for (int j = 0; j <= jmax; ++j) {
    int key0 = j * 32;
    v16bf bk0a, bk0b, bk1a, bk1b;
    {
      const __bf16* kr0 = k + (baseRow + key0 + nloc) * HDIM + col;
      const __bf16* kr1 = k + (baseRow + key0 + 16 + nloc) * HDIM + col;
#pragma unroll
      for (int e = 0; e < 16; ++e) {
        int d = kb + e;
        bk0a[e] = kr0[d];
        bk0b[e] = kr0[32 + d];
        bk1a[e] = kr1[d];
        bk1b[e] = kr1[32 + d];
      }
    }
    v8f s0 = {}, s1 = {};
    s0 = WMMA_BF16(aq0, bk0a, s0);
    s0 = WMMA_BF16(aq1, bk0b, s0);
    s1 = WMMA_BF16(aq0, bk1a, s1);
    s1 = WMMA_BF16(aq1, bk1b, s1);

    float p0[8], p1[8], corr[8];
#pragma unroll
    for (int r = 0; r < 8; ++r) {
      int qi = q0 + rowOff + r;
      float x0 = s0[r] * 0.125f;  // 1/sqrt(64)
      float x1 = s1[r] * 0.125f;
      if (key0 + nloc > qi) x0 = -3.0e38f;
      if (key0 + 16 + nloc > qi) x1 = -3.0e38f;
      float rm = fmaxf(x0, x1);
#pragma unroll
      for (int msk = 1; msk < 16; msk <<= 1) rm = fmaxf(rm, __shfl_xor(rm, msk, 32));
      float nm = fmaxf(run_m[r], rm);
      corr[r] = __expf(run_m[r] - nm);
      run_m[r] = nm;
      x0 = __expf(x0 - nm);
      x1 = __expf(x1 - nm);
      float rs = x0 + x1;
#pragma unroll
      for (int msk = 1; msk < 16; msk <<= 1) rs += __shfl_xor(rs, msk, 32);
      run_l[r] = run_l[r] * corr[r] + rs;
      p0[r] = x0;
      p1[r] = x1;
    }

#pragma unroll
    for (int r = 0; r < 8; ++r) {
      sP[rowOff + r][nloc] = (__bf16)p0[r];
      sP[rowOff + r][16 + nloc] = (__bf16)p1[r];
    }
    __syncthreads();
    v16bf ap;
#pragma unroll
    for (int e = 0; e < 16; ++e) ap[e] = sP[m][a_k_of(lane, e)];
    __syncthreads();

#pragma unroll
    for (int g = 0; g < 4; ++g) {
      v16bf bv;
#pragma unroll
      for (int e = 0; e < 16; ++e)
        bv[e] = v[(baseRow + key0 + kb + e) * HDIM + col + g * 16 + nloc];
      v8f t = oacc[g];
#pragma unroll
      for (int r = 0; r < 8; ++r) t[r] *= corr[r];
      oacc[g] = WMMA_BF16(ap, bv, t);
    }
  }

#pragma unroll
  for (int r = 0; r < 8; ++r) {
    float inv = 1.f / run_l[r];
    long orow = (baseRow + q0 + rowOff + r) * HDIM + col;
#pragma unroll
    for (int g = 0; g < 4; ++g)
      o[orow + g * 16 + nloc] = (__bf16)(oacc[g][r] * inv);
  }
}

// ---------------------------------------------------------------------------
// Host orchestration
// ---------------------------------------------------------------------------
extern "C" void kernel_launch(void* const* d_in, const int* in_sizes, int n_in,
                              void* d_out, int out_size, void* d_ws, size_t ws_size,
                              hipStream_t stream) {
  (void)n_in; (void)out_size; (void)ws_size;
  const int* ids = (const int*)d_in[0];
  const float* tok = (const float*)d_in[1];
  const float* pos = (const float*)d_in[2];
  const float* Wq = (const float*)d_in[3];
  const float* bq = (const float*)d_in[4];
  const float* Wk = (const float*)d_in[5];
  const float* bk = (const float*)d_in[6];
  const float* Wv = (const float*)d_in[7];
  const float* bv = (const float*)d_in[8];
  const float* Wo = (const float*)d_in[9];
  const float* bo = (const float*)d_in[10];
  const float* ln1w = (const float*)d_in[11];
  const float* ln1b = (const float*)d_in[12];
  const float* ln2w = (const float*)d_in[13];
  const float* ln2b = (const float*)d_in[14];
  const float* Wup = (const float*)d_in[15];
  const float* bup = (const float*)d_in[16];
  const float* Wdown = (const float*)d_in[17];
  const float* bdown = (const float*)d_in[18];
  const float* lnfw = (const float*)d_in[19];
  const float* lnfb = (const float*)d_in[20];

  const int B = 2, T = 1024, H = HDIM, L = 12, I = 4 * HDIM;
  const int M = B * T;
  const int V = in_sizes[1] / H;  // 50257

  // Workspace carve-up (~35 MB): fp32 residual + bf16 operands.
  char* w = (char*)d_ws;
  float* x = (float*)w;        w += (size_t)M * H * sizeof(float);
  __bf16* h = (__bf16*)w;      w += (size_t)M * H * sizeof(__bf16);
  __bf16* qb = (__bf16*)w;     w += (size_t)M * H * sizeof(__bf16);
  __bf16* kbuf = (__bf16*)w;   w += (size_t)M * H * sizeof(__bf16);
  __bf16* vb = (__bf16*)w;     w += (size_t)M * H * sizeof(__bf16);
  __bf16* ob = (__bf16*)w;     w += (size_t)M * H * sizeof(__bf16);
  __bf16* ub = (__bf16*)w;     w += (size_t)M * I * sizeof(__bf16);

  swarm_embed_kernel<<<M, 256, 0, stream>>>(ids, tok, pos, x, T);

  dim3 gLN(M / 8);
  dim3 gH(H / 128, M / 64);   // (6, 32)
  dim3 gI(I / 128, M / 64);   // (24, 32)
  dim3 gA(T / 16, NHEADS, B);

  for (int l = 0; l < L; ++l) {
    long oHH = (long)l * H * H;
    long oH = (long)l * H;
    long oIH = (long)l * I * H;
    long oI = (long)l * I;

    swarm_ln_kernel<<<gLN, 256, 0, stream>>>(x, ln1w + oH, ln1b + oH, h, M);
    swarm_gemm_kernel<<<gH, 256, 0, stream>>>(h, Wq + oHH, bq + oH, nullptr, qb,   M, H, H, GF_WBF16);
    swarm_gemm_kernel<<<gH, 256, 0, stream>>>(h, Wk + oHH, bk + oH, nullptr, kbuf, M, H, H, GF_WBF16);
    swarm_gemm_kernel<<<gH, 256, 0, stream>>>(h, Wv + oHH, bv + oH, nullptr, vb,   M, H, H, GF_WBF16);
    swarm_attn_kernel<<<gA, 32, 0, stream>>>(qb, kbuf, vb, ob, T);
    swarm_gemm_kernel<<<gH, 256, 0, stream>>>(ob, Wo + oHH, bo + oH, x, nullptr, M, H, H, GF_ADDRES);

    swarm_ln_kernel<<<gLN, 256, 0, stream>>>(x, ln2w + oH, ln2b + oH, h, M);
    swarm_gemm_kernel<<<gI, 256, 0, stream>>>(h, Wup + oIH, bup + oI, nullptr, ub, M, I, H, GF_GELU | GF_WBF16);
    swarm_gemm_kernel<<<gH, 256, 0, stream>>>(ub, Wdown + oIH, bdown + oH, x, nullptr, M, H, I, GF_ADDRES);
  }

  swarm_ln_kernel<<<gLN, 256, 0, stream>>>(x, lnfw, lnfb, h, M);
  dim3 gV((V + 127) / 128, M / 64);  // (393, 32) — tied lm_head, tok_emb L2-resident
  swarm_gemm_kernel<<<gV, 256, 0, stream>>>(h, tok, nullptr, (float*)d_out, nullptr, M, V, H, GF_WF32);
}
